// Mamba2_90314572300646
// MI455X (gfx1250) — compile-verified, matrix operand order checked by reading, not run on hardware
//
#include <hip/hip_runtime.h>
#include <hip/hip_bf16.h>

// ---------------- Problem constants (from reference) ----------------
#define D_MODEL   1024
#define D_INNER   2048
#define D_STATE   64
#define D_CONV    4
#define HEADDIM   64
#define NHEADS    32
#define D_CONV_CH 2176            // D_INNER + 2*D_STATE
#define D_IN_PROJ 4256            // 2*D_INNER + 2*D_STATE + NHEADS
#define BSZ       2
#define SEQL      4096
#define MROWS     (BSZ * SEQL)    // 8192

// ---------------- WMMA types ----------------
typedef __attribute__((ext_vector_type(16))) __bf16 v16bf;
typedef __attribute__((ext_vector_type(8)))  float  v8f;

union Frag {
    v16bf v;
    uint4 q[2];
};

__device__ __forceinline__ unsigned short f2bf_rne(float f) {
    unsigned int u = __float_as_uint(f);
    unsigned int lsb = (u >> 16) & 1u;
    u += 0x7fffu + lsb;
    return (unsigned short)(u >> 16);
}

__device__ __forceinline__ float sigmoidf_fast(float x) {
    return 1.0f / (1.0f + __expf(-x));
}

// ---------------- f32 -> bf16 cast (plain) ----------------
__global__ void cast_bf16_kernel(const float* __restrict__ src,
                                 unsigned short* __restrict__ dst, long n) {
    long i = (long)blockIdx.x * blockDim.x + threadIdx.x;
    if (i < n) dst[i] = f2bf_rne(src[i]);
}

// ---------------- f32 [K,N] -> bf16 transposed [N,K] ----------------
// One-time weight prep so the GEMM can stage B with vectorized b128 copies.
__global__ void cast_bf16_T_kernel(const float* __restrict__ src,
                                   unsigned short* __restrict__ dst,
                                   int K, int N) {
    long i = (long)blockIdx.x * blockDim.x + threadIdx.x;
    long total = (long)K * N;
    if (i >= total) return;
    int n = (int)(i / K);
    int k = (int)(i - (long)n * K);
    dst[i] = f2bf_rne(src[(size_t)k * N + n]);
}

// ---------------- WMMA GEMM: C[M,N] f32 = A[M,K] bf16 * BT[N,K] bf16 ----------------
// Block tile 128x128, K-step 32, double-buffered LDS. 256 threads = 8 waves in
// 4(M) x 2(N); each wave owns 32x64 => 8 v_wmma_f32_16x16x32_bf16 per k-step.
// Assumes M % 128 == 0, K % 32 == 0 (true for both calls); N tail guarded per row.
#define GBM 128
#define GBN 128
#define GBK 32
#define LDSS 48   // padded row stride (elements); 96 bytes => rows stay 16B-aligned

__device__ __forceinline__ void stage_tiles(const unsigned short* __restrict__ A,
                                            const unsigned short* __restrict__ BT,
                                            unsigned short* __restrict__ sA,
                                            unsigned short* __restrict__ sB,
                                            int m0, int n0, int k0,
                                            int N, int K, int tid) {
    const int row = tid >> 1;           // 0..127
    const int col = (tid & 1) << 4;     // 0 or 16 elements
    {
        const unsigned short* g = A + (size_t)(m0 + row) * K + k0 + col;
        *(uint4*)&sA[row * LDSS + col]     = *(const uint4*)(g);
        *(uint4*)&sA[row * LDSS + col + 8] = *(const uint4*)(g + 8);
    }
    {
        uint4 q0 = {0, 0, 0, 0}, q1 = {0, 0, 0, 0};
        if (n0 + row < N) {
            const unsigned short* g = BT + (size_t)(n0 + row) * K + k0 + col;
            q0 = *(const uint4*)(g);
            q1 = *(const uint4*)(g + 8);
        }
        *(uint4*)&sB[row * LDSS + col]     = q0;
        *(uint4*)&sB[row * LDSS + col + 8] = q1;
    }
}

__global__ __launch_bounds__(256)
void gemm_bf16_wmma(const unsigned short* __restrict__ A,
                    const unsigned short* __restrict__ BT,
                    float* __restrict__ C,
                    int M, int N, int K) {
    __shared__ __align__(16) unsigned short sA[2][GBM * LDSS];
    __shared__ __align__(16) unsigned short sB[2][GBN * LDSS];

    const int tid  = threadIdx.x;
    const int lane = tid & 31;
    const int wave = tid >> 5;
    const int wm   = wave >> 1;   // 0..3
    const int wn   = wave & 1;    // 0..1
    const int m0   = blockIdx.y * GBM;
    const int n0   = blockIdx.x * GBN;

    v8f acc[2][4];
    #pragma unroll
    for (int mi = 0; mi < 2; ++mi)
        #pragma unroll
        for (int ni = 0; ni < 4; ++ni)
            #pragma unroll
            for (int r = 0; r < 8; ++r)
                acc[mi][ni][r] = 0.0f;

    const int koff = (lane < 16) ? 0 : 8;   // ISA 16-bit A 16x32 lane layout
    const int rsel = lane & 15;

    stage_tiles(A, BT, sA[0], sB[0], m0, n0, 0, N, K, tid);
    __syncthreads();

    int buf = 0;
    for (int k0 = 0; k0 < K; k0 += GBK) {
        const int nbuf = buf ^ 1;
        if (k0 + GBK < K) {
            stage_tiles(A, BT, sA[nbuf], sB[nbuf], m0, n0, k0 + GBK, N, K, tid);
            if (k0 + 2 * GBK < K)
                __builtin_prefetch(A + (size_t)(m0 + (tid >> 1)) * K + k0 + 2 * GBK, 0, 1);
        }

        Frag fa[2], fb[4];
        #pragma unroll
        for (int mi = 0; mi < 2; ++mi) {
            const unsigned short* p = &sA[buf][(wm * 32 + mi * 16 + rsel) * LDSS + koff];
            fa[mi].q[0] = *(const uint4*)(p);
            fa[mi].q[1] = *(const uint4*)(p + 16);
        }
        #pragma unroll
        for (int ni = 0; ni < 4; ++ni) {
            const unsigned short* p = &sB[buf][(wn * 64 + ni * 16 + rsel) * LDSS + koff];
            fb[ni].q[0] = *(const uint4*)(p);
            fb[ni].q[1] = *(const uint4*)(p + 16);
        }
        #pragma unroll
        for (int mi = 0; mi < 2; ++mi)
            #pragma unroll
            for (int ni = 0; ni < 4; ++ni)
                acc[mi][ni] = __builtin_amdgcn_wmma_f32_16x16x32_bf16(
                    false, fa[mi].v, false, fb[ni].v,
                    (short)0, acc[mi][ni], false, false);

        __syncthreads();
        buf = nbuf;
    }

    // writeback: C/D 16x16 layout: n = lane&15, m = r + (lane>=16 ? 8 : 0)
    const int mhi = (lane >> 4) << 3;
    #pragma unroll
    for (int mi = 0; mi < 2; ++mi)
        #pragma unroll
        for (int ni = 0; ni < 4; ++ni) {
            int nn = n0 + wn * 64 + ni * 16 + rsel;
            if (nn >= N) continue;
            #pragma unroll
            for (int r = 0; r < 8; ++r) {
                int mm = m0 + wm * 32 + mi * 16 + r + mhi;
                C[(size_t)mm * N + nn] = acc[mi][ni][r];
            }
        }
}

// ---------------- dt = softplus(dt_raw + bias) ----------------
__global__ void dt_softplus_kernel(const float* __restrict__ zx,
                                   const float* __restrict__ bias,
                                   float* __restrict__ dt) {
    int i = blockIdx.x * blockDim.x + threadIdx.x;
    if (i >= MROWS * NHEADS) return;
    int r = i >> 5, h = i & 31;
    float v = zx[(size_t)r * D_IN_PROJ + (D_INNER + D_CONV_CH) + h] + bias[h];
    dt[i] = (v > 15.0f) ? v : log1pf(__expf(v));
}

// ---------------- causal depthwise conv(width 4) + bias + SiLU ----------------
__global__ void conv_silu_kernel(const float* __restrict__ zx,
                                 const float* __restrict__ w,
                                 const float* __restrict__ bvec,
                                 float* __restrict__ out) {
    long i = (long)blockIdx.x * blockDim.x + threadIdx.x;
    const long total = (long)MROWS * D_CONV_CH;
    if (i >= total) return;
    int  c   = (int)(i % D_CONV_CH);
    long row = i / D_CONV_CH;
    int  l   = (int)(row % SEQL);
    long bL  = row - l;                 // b*L
    float acc = bvec[c];
    #pragma unroll
    for (int j = 0; j < D_CONV; ++j) {
        int lt = l - (D_CONV - 1) + j;
        if (lt >= 0)
            acc += w[c * D_CONV + j] * zx[(bL + lt) * D_IN_PROJ + D_INNER + c];
    }
    out[i] = acc * sigmoidf_fast(acc);
}

// ---------------- sequential SSM scan over L ----------------
// One block per (batch, head). Thread p owns state row s[p, 0..63] in VGPRs.
__global__ __launch_bounds__(64)
void scan_kernel(const float* __restrict__ xBC,   // [MROWS, 2176] post conv+silu
                 const float* __restrict__ dtsp,  // [MROWS, 32]
                 const float* __restrict__ A_log,
                 const float* __restrict__ Dvec,
                 float* __restrict__ y_ssm) {     // [MROWS, 2048]
    const int b = blockIdx.x >> 5;
    const int h = blockIdx.x & 31;
    const int p = threadIdx.x;        // 0..63 (headdim index)
    __shared__ float sB[D_STATE], sC[D_STATE];

    float s[D_STATE];
    #pragma unroll
    for (int n = 0; n < D_STATE; ++n) s[n] = 0.0f;

    const float Acoef = -__expf(A_log[h]);
    const float Dh    = Dvec[h];

    for (int t = 0; t < SEQL; ++t) {
        const size_t row = (size_t)b * SEQL + t;
        const float* base = xBC + row * D_CONV_CH;
        sB[p] = base[D_INNER + p];
        sC[p] = base[D_INNER + D_STATE + p];
        const float xv  = base[h * HEADDIM + p];
        const float dtv = dtsp[row * NHEADS + h];
        const float dA  = __expf(dtv * Acoef);
        const float cf  = dtv * xv;
        __syncthreads();
        float y = 0.0f;
        #pragma unroll
        for (int n = 0; n < D_STATE; ++n) {
            s[n] = s[n] * dA + cf * sB[n];
            y += s[n] * sC[n];
        }
        y_ssm[row * D_INNER + h * HEADDIM + p] = y + Dh * xv;
        __syncthreads();
    }
}

// ---------------- y * silu(z), RMSNorm, cast to bf16 ----------------
__global__ __launch_bounds__(256)
void gate_norm_kernel(const float* __restrict__ y_ssm,
                      const float* __restrict__ zx,
                      const float* __restrict__ w,
                      unsigned short* __restrict__ out_b) {
    const int row = blockIdx.x;
    const int tid = threadIdx.x;
    float g[8];
    float ss = 0.0f;
    #pragma unroll
    for (int j = 0; j < 8; ++j) {
        int c = j * 256 + tid;
        float zv = zx[(size_t)row * D_IN_PROJ + c];
        float gv = y_ssm[(size_t)row * D_INNER + c] * zv * sigmoidf_fast(zv);
        g[j] = gv;
        ss += gv * gv;
    }
    #pragma unroll
    for (int o = 16; o > 0; o >>= 1) ss += __shfl_xor(ss, o, 32);
    __shared__ float red[8];
    if ((tid & 31) == 0) red[tid >> 5] = ss;
    __syncthreads();
    float tot = 0.0f;
    #pragma unroll
    for (int i = 0; i < 8; ++i) tot += red[i];
    const float scale = rsqrtf(tot / (float)D_INNER + 1e-5f);
    #pragma unroll
    for (int j = 0; j < 8; ++j) {
        int c = j * 256 + tid;
        out_b[(size_t)row * D_INNER + c] = f2bf_rne(g[j] * scale * w[c]);
    }
}

// ---------------- host launch ----------------
extern "C" void kernel_launch(void* const* d_in, const int* in_sizes, int n_in,
                              void* d_out, int out_size, void* d_ws, size_t ws_size,
                              hipStream_t stream) {
    const float* x        = (const float*)d_in[0];
    const float* W_in     = (const float*)d_in[1];
    const float* conv_w   = (const float*)d_in[2];
    const float* conv_b   = (const float*)d_in[3];
    const float* dt_bias  = (const float*)d_in[4];
    const float* A_log    = (const float*)d_in[5];
    const float* Dv       = (const float*)d_in[6];
    const float* norm_w   = (const float*)d_in[7];
    const float* W_out    = (const float*)d_in[8];
    float* out = (float*)d_out;

    char* ws = (char*)d_ws;
    size_t off = 0;
    auto take = [&](size_t bytes) -> void* {
        void* p = ws + off;
        off += (bytes + 255) & ~(size_t)255;
        return p;
    };
    unsigned short* xb    = (unsigned short*)take((size_t)MROWS * D_MODEL * 2);
    unsigned short* WibT  = (unsigned short*)take((size_t)D_MODEL * D_IN_PROJ * 2); // [4256][1024]
    unsigned short* WobT  = (unsigned short*)take((size_t)D_INNER * D_MODEL * 2);   // [1024][2048]
    float* zx             = (float*)take((size_t)MROWS * D_IN_PROJ * 4);
    float* xbc            = (float*)take((size_t)MROWS * D_CONV_CH * 4);
    float* dts            = (float*)take((size_t)MROWS * NHEADS * 4);
    float* yssm           = (float*)take((size_t)MROWS * D_INNER * 4);
    unsigned short* yb    = (unsigned short*)take((size_t)MROWS * D_INNER * 2);

    // 1) casts to bf16 (weights also transposed to [N][K] for vectorized staging)
    {
        long n1 = (long)MROWS * D_MODEL;
        cast_bf16_kernel<<<(unsigned)((n1 + 255) / 256), 256, 0, stream>>>(x, xb, n1);
        long n2 = (long)D_MODEL * D_IN_PROJ;
        cast_bf16_T_kernel<<<(unsigned)((n2 + 255) / 256), 256, 0, stream>>>(
            W_in, WibT, D_MODEL, D_IN_PROJ);
        long n3 = (long)D_INNER * D_MODEL;
        cast_bf16_T_kernel<<<(unsigned)((n3 + 255) / 256), 256, 0, stream>>>(
            W_out, WobT, D_INNER, D_MODEL);
    }

    // 2) in_proj GEMM: [8192x1024] x [1024x4256] -> zx
    {
        dim3 grid((D_IN_PROJ + GBN - 1) / GBN, MROWS / GBM);
        gemm_bf16_wmma<<<grid, 256, 0, stream>>>(xb, WibT, zx, MROWS, D_IN_PROJ, D_MODEL);
    }

    // 3) softplus(dt)
    dt_softplus_kernel<<<(MROWS * NHEADS + 255) / 256, 256, 0, stream>>>(zx, dt_bias, dts);

    // 4) depthwise conv + SiLU
    {
        long tot = (long)MROWS * D_CONV_CH;
        conv_silu_kernel<<<(unsigned)((tot + 255) / 256), 256, 0, stream>>>(zx, conv_w, conv_b, xbc);
    }

    // 5) sequential SSM scan
    scan_kernel<<<BSZ * NHEADS, HEADDIM, 0, stream>>>(xbc, dts, A_log, Dv, yssm);

    // 6) gate + RMSNorm -> bf16 activations
    gate_norm_kernel<<<MROWS, 256, 0, stream>>>(yssm, zx, norm_w, yb);

    // 7) out_proj GEMM: [8192x2048] x [2048x1024] -> out (f32)
    {
        dim3 grid(D_MODEL / GBN, MROWS / GBM);
        gemm_bf16_wmma<<<grid, 256, 0, stream>>>(yb, WobT, out, MROWS, D_MODEL, D_INNER);
    }
}
